// RGCN_26972394619139
// MI455X (gfx1250) — matrix operand order repaired and checked by reading, not compiled, not run on hardware
//
#include <hip/hip_runtime.h>

typedef __attribute__((ext_vector_type(16))) _Float16 v16h;
typedef __attribute__((ext_vector_type(8)))  _Float16 v8h;
typedef __attribute__((ext_vector_type(8)))  float    v8f;

#define NNODE 50000
#define NEDGE 800000
#define DIM   256
#define NREL  8
#define NBAS  4
#define NMAT  9                      // 8 relations + root
#define MTILES 3125                  // 50000 / 16 (exact)
#define KTILES 8                     // 256 / 32
#define OTILES 16                    // 256 / 16
#define XPK_HALVES (MTILES*KTILES*512)        // 12,800,000 f16
#define WPK_HALVES (NMAT*OTILES*KTILES*512)   // 589,824 f16 per layer

// ---------------------------------------------------------------------------
// Pack x (f32 [N,256]) into WMMA A-fragment order (16x32 f16 tiles).
// A layout (ISA 7.12.2): lane = m + 16*((kk>>3)&1); e = (kk&7) + 8*(kk>>4).
// Packed index p = ((mt*8 + kt)*32 + lane)*16 + e  -> one b128 pair per lane.
// ---------------------------------------------------------------------------
__global__ void __launch_bounds__(256)
pack_x_kernel(const float* __restrict__ xin, _Float16* __restrict__ xpk) {
  int p    = blockIdx.x * 256 + threadIdx.x;        // exact: XPK_HALVES/256 blocks
  int e    = p & 15;
  int lane = (p >> 4) & 31;
  int kt   = (p >> 9) & 7;
  int mt   = p >> 12;
  int m    = lane & 15;
  int half = lane >> 4;
  int kk   = (e & 7) + (half << 3) + ((e >> 3) << 4);   // inverse of A mapping
  int n    = mt * 16 + m;
  int k    = kt * 32 + kk;
  xpk[p] = (_Float16)xin[n * DIM + k];
}

// ---------------------------------------------------------------------------
// Build per-relation weights W_r = sum_b comp[r,b]*basis[b]  (+ root as j==8)
// for both layers, packed in WMMA B-fragment order (32x16 f16 tiles).
// B layout: lane = (o&15) + 16*((kk>>3)&1); e = (kk&7) + 8*(kk>>4).
// Packed index q = (((j*16 + ot)*8 + kt)*32 + lane)*16 + e.
// ---------------------------------------------------------------------------
__global__ void __launch_bounds__(256)
pack_w_kernel(const float* __restrict__ b1, const float* __restrict__ c1,
              const float* __restrict__ r1, const float* __restrict__ b2,
              const float* __restrict__ c2, const float* __restrict__ r2,
              _Float16* __restrict__ wpk) {
  int p     = blockIdx.x * 256 + threadIdx.x;       // exact: 2*WPK_HALVES/256
  int layer = p >= WPK_HALVES;
  int q     = p - layer * WPK_HALVES;
  int e     = q & 15;
  int lane  = (q >> 4) & 31;
  int kt    = (q >> 9) & 7;
  int ot    = (q >> 12) & 15;
  int j     = q >> 16;                              // 0..8
  int o     = ot * 16 + (lane & 15);
  int k     = kt * 32 + (e & 7) + (lane >> 4) * 8 + ((e >> 3) << 4);
  const float* basis = layer ? b2 : b1;
  const float* comp  = layer ? c2 : c1;
  const float* root  = layer ? r2 : r1;
  float v;
  if (j < NREL) {
    v = 0.f;
#pragma unroll
    for (int b = 0; b < NBAS; ++b)
      v += comp[j * NBAS + b] * basis[b * DIM * DIM + k * DIM + o];
  } else {
    v = root[k * DIM + o];
  }
  wpk[p] = (_Float16)v;
}

// ---------------------------------------------------------------------------
// Per-(dst, relation) edge counts, then in-place 1/max(cnt,1).
// ---------------------------------------------------------------------------
__global__ void __launch_bounds__(256)
count_kernel(const int* __restrict__ dst, const int* __restrict__ et,
             unsigned* __restrict__ cnt) {
  int e = blockIdx.x * 256 + threadIdx.x;           // exact: NEDGE/256 blocks
  atomicAdd(&cnt[dst[e] * NREL + et[e]], 1u);
}

__global__ void __launch_bounds__(256)
norm_kernel(unsigned* __restrict__ cnt) {
  int i = blockIdx.x * 256 + threadIdx.x;
  if (i >= NNODE * NREL) return;
  unsigned c = cnt[i];
  ((float*)cnt)[i] = 1.0f / (float)(c ? c : 1u);
}

// ---------------------------------------------------------------------------
// Batched GEMM: xr[n, j, :] = x[n, :] @ W_j  (9 matrices, K = 256), f16 out.
// One wave per 16x64 output tile: 4 f32 accumulators, 32 v_wmma per wave.
// grid = (ceil(MTILES/8), NMAT, 4), block = 256 (8 waves).
// ---------------------------------------------------------------------------
__global__ void __launch_bounds__(256)
rgcn_gemm_kernel(const _Float16* __restrict__ xpk,
                 const _Float16* __restrict__ wpk,
                 _Float16* __restrict__ xr) {
  int lane = threadIdx.x & 31;
  int mt   = blockIdx.x * 8 + (threadIdx.x >> 5);   // wave-uniform -> full EXEC
  if (mt >= MTILES) return;
  int j  = blockIdx.y;                              // matrix 0..8
  int oq = blockIdx.z;                              // 64-wide column quad

  const _Float16* ap = xpk + (size_t)mt * (KTILES * 512) + lane * 16;
  const _Float16* bp = wpk + ((size_t)(j * OTILES + oq * 4) * KTILES) * 512 + lane * 16;

  v8f acc0 = {}, acc1 = {}, acc2 = {}, acc3 = {};
#pragma unroll
  for (int kt = 0; kt < KTILES; ++kt) {
    v16h a  = *(const v16h*)(ap + kt * 512);
    v16h b0 = *(const v16h*)(bp + (0 * KTILES + kt) * 512);
    v16h b1 = *(const v16h*)(bp + (1 * KTILES + kt) * 512);
    v16h b2 = *(const v16h*)(bp + (2 * KTILES + kt) * 512);
    v16h b3 = *(const v16h*)(bp + (3 * KTILES + kt) * 512);
    acc0 = __builtin_amdgcn_wmma_f32_16x16x32_f16(false, a, false, b0, (short)0, acc0, false, false);
    acc1 = __builtin_amdgcn_wmma_f32_16x16x32_f16(false, a, false, b1, (short)0, acc1, false, false);
    acc2 = __builtin_amdgcn_wmma_f32_16x16x32_f16(false, a, false, b2, (short)0, acc2, false, false);
    acc3 = __builtin_amdgcn_wmma_f32_16x16x32_f16(false, a, false, b3, (short)0, acc3, false, false);
  }

  // D fragment: VGPR jv -> (M = jv + 8*(lane>=16), N = lane&15)
  int nbase = mt * 16 + 8 * (lane >> 4);
  int ncol  = lane & 15;
  v8f accs[4] = {acc0, acc1, acc2, acc3};
#pragma unroll
  for (int t = 0; t < 4; ++t) {
    int o = (oq * 4 + t) * 16 + ncol;
#pragma unroll
    for (int jv = 0; jv < 8; ++jv) {
      int n = nbase + jv;
      xr[((size_t)n * NMAT + j) * DIM + o] = (_Float16)accs[t][jv];
    }
  }
}

// ---------------------------------------------------------------------------
// Edge pass: one wave per edge. Gather xr[src, rel] (256 f16 = 16B/lane),
// scale by 1/cnt(dst,rel), atomic-add into agg[dst] (L2-resident, 51 MB).
// ---------------------------------------------------------------------------
__global__ void __launch_bounds__(256)
scatter_kernel(const _Float16* __restrict__ xr, const float* __restrict__ norm,
               const int* __restrict__ src, const int* __restrict__ dst,
               const int* __restrict__ et, float* __restrict__ agg) {
  int lane = threadIdx.x & 31;
  int e    = blockIdx.x * 8 + (threadIdx.x >> 5);   // exact: NEDGE/8 blocks
  int s = src[e], d = dst[e], r = et[e];
  float nm = norm[d * NREL + r];
  v8h row = *(const v8h*)(xr + ((size_t)s * NMAT + r) * DIM + lane * 8);
  float* out = agg + (size_t)d * DIM + lane * 8;
#pragma unroll
  for (int i = 0; i < 8; ++i)
    __hip_atomic_fetch_add(&out[i], nm * (float)row[i],
                           __ATOMIC_RELAXED, __HIP_MEMORY_SCOPE_AGENT);
}

// ---------------------------------------------------------------------------
// out = [relu](agg + x@root + bias); root product sits at xr[:, 8, :].
// ---------------------------------------------------------------------------
__global__ void __launch_bounds__(256)
combine_kernel(const float* __restrict__ agg, const _Float16* __restrict__ xr,
               const float* __restrict__ bias, float* __restrict__ out, int do_relu) {
  int i = blockIdx.x * 256 + threadIdx.x;           // exact: N*DIM/256 blocks
  int n = i >> 8, o = i & 255;
  float v = agg[i] + (float)xr[((size_t)n * NMAT + 8) * DIM + o] + bias[o];
  out[i] = do_relu ? fmaxf(v, 0.f) : v;
}

// ---------------------------------------------------------------------------
extern "C" void kernel_launch(void* const* d_in, const int* in_sizes, int n_in,
                              void* d_out, int out_size, void* d_ws, size_t ws_size,
                              hipStream_t stream) {
  const float* x      = (const float*)d_in[0];
  const int*   eidx   = (const int*)d_in[1];
  const int*   etype  = (const int*)d_in[2];
  const float* basis1 = (const float*)d_in[3];
  const float* comp1  = (const float*)d_in[4];
  const float* root1  = (const float*)d_in[5];
  const float* bias1  = (const float*)d_in[6];
  const float* basis2 = (const float*)d_in[7];
  const float* comp2  = (const float*)d_in[8];
  const float* root2  = (const float*)d_in[9];
  const float* bias2  = (const float*)d_in[10];
  const int* src = eidx;
  const int* dst = eidx + NEDGE;
  (void)in_sizes; (void)n_in; (void)out_size; (void)ws_size;

  char* ws = (char*)d_ws;
  size_t off = 0;
  _Float16* xpk = (_Float16*)(ws + off); off += (size_t)XPK_HALVES * 2;       // 25.6 MB
  _Float16* wpk = (_Float16*)(ws + off); off += (size_t)2 * WPK_HALVES * 2;   //  2.4 MB
  unsigned* cnt = (unsigned*)(ws + off); off += (size_t)NNODE * NREL * 4;     //  1.6 MB
  _Float16* xr  = (_Float16*)(ws + off); off += (size_t)NNODE * NMAT * DIM * 2; // 230.4 MB
  float*    agg = (float*)(ws + off);    off += (size_t)NNODE * DIM * 4;      // 51.2 MB
  float*    h   = (float*)(ws + off);    off += (size_t)NNODE * DIM * 4;      // 51.2 MB

  hipMemsetAsync(cnt, 0, (size_t)NNODE * NREL * 4, stream);
  hipMemsetAsync(agg, 0, (size_t)NNODE * DIM * 4, stream);

  count_kernel<<<NEDGE / 256, 256, 0, stream>>>(dst, etype, cnt);
  norm_kernel<<<(NNODE * NREL + 255) / 256, 256, 0, stream>>>(cnt);
  pack_w_kernel<<<2 * WPK_HALVES / 256, 256, 0, stream>>>(basis1, comp1, root1,
                                                          basis2, comp2, root2, wpk);
  dim3 ggrid((MTILES + 7) / 8, NMAT, 4);

  // ---- layer 1 ----
  pack_x_kernel<<<XPK_HALVES / 256, 256, 0, stream>>>(x, xpk);
  rgcn_gemm_kernel<<<ggrid, 256, 0, stream>>>(xpk, wpk, xr);
  scatter_kernel<<<NEDGE / 8, 256, 0, stream>>>(xr, (const float*)cnt, src, dst, etype, agg);
  combine_kernel<<<NNODE * DIM / 256, 256, 0, stream>>>(agg, xr, bias1, h, 1);

  // ---- layer 2 ----
  hipMemsetAsync(agg, 0, (size_t)NNODE * DIM * 4, stream);
  pack_x_kernel<<<XPK_HALVES / 256, 256, 0, stream>>>(h, xpk);
  rgcn_gemm_kernel<<<ggrid, 256, 0, stream>>>(xpk, wpk + WPK_HALVES, xr);
  scatter_kernel<<<NEDGE / 8, 256, 0, stream>>>(xr, (const float*)cnt, src, dst, etype, agg);
  combine_kernel<<<NNODE * DIM / 256, 256, 0, stream>>>(agg, xr, bias2, (float*)d_out, 0);
}
